// MT_RNN_27582279975304
// MI455X (gfx1250) — matrix operand order repaired and checked by reading, not compiled, not run on hardware
//
#include <hip/hip_runtime.h>
#include <hip/hip_bf16.h>
#include <math.h>

// ---------------------------------------------------------------------------
// MT_RNN on gfx1250: bf16 WMMA (v_wmma_f32_16x16x32_bf16) everywhere.
//   Stage 1: feat  = tanh(tanh(x@Wx0^T+bx0)@Wx1^T+bx1)     (parallel GEMM)
//   Stage 2: h-scan (serial over T, parallel over B tiles; h lives in LDS,
//            feat tiles double-buffered into LDS with async-to-LDS loads)
//   Stage 3: y_t = MLP(h_t)@Wg^T+bg over all (t,b) at once (parallel GEMM,
//            h rows staged to LDS with async-to-LDS loads)
// ---------------------------------------------------------------------------

typedef __attribute__((ext_vector_type(16))) __bf16       v16bf;
typedef __attribute__((ext_vector_type(8)))  float        v8f;
typedef __attribute__((ext_vector_type(4)))  unsigned int v4u;
typedef __attribute__((ext_vector_type(8)))  unsigned int v8u;

#define T_LEN 1024
#define B_DIM 256
#define XD    64
#define H_DIM 256
#define F_DIM 256

static __device__ __forceinline__ unsigned short f2bf(float f) {
  unsigned int u = __float_as_uint(f);
  u += 0x7FFFu + ((u >> 16) & 1u);          // round-to-nearest-even
  return (unsigned short)(u >> 16);
}

// Branch-free tanh: one v_exp_f32 (TRANS, co-issues with WMMA) + ~5 VALU.
// tanh(x) = sign(x) * (1 - 2/(e^{2|x|}+1)); large |x| -> exp=inf -> rcp=0 -> 1.
static __device__ __forceinline__ float fast_tanh(float x) {
  const float ax = __builtin_fabsf(x);
  const float e  = __builtin_amdgcn_exp2f(ax * 2.885390081777927f); // e^{2|x|}
  const float r  = 1.0f - 2.0f * __builtin_amdgcn_rcpf(e + 1.0f);
  return __builtin_copysignf(r, x);
}

static __device__ __forceinline__ v16bf make_frag(v4u a, v4u b) {
  v8u t;
  t[0] = a[0]; t[1] = a[1]; t[2] = a[2]; t[3] = a[3];
  t[4] = b[0]; t[5] = b[1]; t[6] = b[2]; t[7] = b[3];
  return __builtin_bit_cast(v16bf, t);
}

// A fragment, 16x32 bf16, from row-major [rows][ld] activations.
// Lane L: row m0+(L&15); K = k0 + (L>>4)*8 + {0..7} and +16..+23  (ISA 7.12.2)
static __device__ __forceinline__ v16bf load_afrag(const unsigned short* base,
                                                   int ld, int m0, int k0, int lane) {
  const unsigned short* p =
      base + (size_t)(m0 + (lane & 15)) * ld + (k0 + ((lane >> 4) << 3));
  return make_frag(*(const v4u*)p, *(const v4u*)(p + 16));
}

// B fragment, 32x16 bf16, from row-major weight W[N][K] (out = in @ W^T).
// Lane L: col n0+(L&15); K = k0 + (L>>4)*8 + {0..7} and +16..+23 (matches A).
static __device__ __forceinline__ v16bf load_bfrag(const unsigned short* W,
                                                   int ld, int n0, int k0, int lane) {
  const unsigned short* p =
      W + (size_t)(n0 + (lane & 15)) * ld + (k0 + ((lane >> 4) << 3));
  return make_frag(*(const v4u*)p, *(const v4u*)(p + 16));
}

static __device__ __forceinline__ v8f wmma_bf(v16bf a, v16bf b, v8f c) {
  return __builtin_amdgcn_wmma_f32_16x16x32_bf16(false, a, false, b, (short)0, c,
                                                 false, false);
}

// --- gfx1250 async-to-LDS (ASYNCcnt-tracked) -------------------------------
// Flat LDS aperture keeps the byte offset in addr[31:0] (ISA 10.2), so the
// low 32 bits of a generic pointer to LDS are the DS destination address.
static __device__ __forceinline__ unsigned lds_off(const void* p) {
  return (unsigned)(unsigned long long)p;
}
static __device__ __forceinline__ void async_ld_b128(unsigned dsoff,
                                                     const void* g) {
  asm volatile("global_load_async_to_lds_b128 %0, %1, off"
               :: "v"(dsoff), "v"(g) : "memory");
}
static __device__ __forceinline__ void wait_async0() {
  asm volatile("s_wait_asynccnt 0x0" ::: "memory");
}

// --------------------------- small helper kernels --------------------------
__global__ void cvt_kernel(const float* __restrict__ s,
                           unsigned short* __restrict__ d, int n) {
  int i = blockIdx.x * 256 + threadIdx.x;
  if (i < n) d[i] = f2bf(s[i]);
}

__global__ void zero16_kernel(unsigned short* __restrict__ d, int n) {
  int i = blockIdx.x * 256 + threadIdx.x;
  if (i < n) d[i] = 0;
}

// --------------------------- stage 1: feature MLP ---------------------------
// 2048 blocks x 256 threads; each block owns 128 rows of [T*B, XD].
__global__ __launch_bounds__(256) void feat_kernel(
    const float* __restrict__ x,
    const unsigned short* __restrict__ Wx0b, const float* __restrict__ bx0,
    const unsigned short* __restrict__ Wx1b, const float* __restrict__ bx1,
    unsigned short* __restrict__ featb) {
  extern __shared__ __align__(16) unsigned short smem[];
  unsigned short* xs   = smem;             // [128][64]  bf16
  unsigned short* act1 = smem + 128 * XD;  // [128][256] bf16

  const int tid = threadIdx.x, lane = tid & 31, wave = tid >> 5;
  const size_t rowBase = (size_t)blockIdx.x * 128;
  const int m0 = wave * 16;
  const int n_l = lane & 15, rb = (lane >> 4) << 3;

  for (int i = tid; i < 128 * XD; i += 256)
    xs[i] = f2bf(x[rowBase * XD + i]);
  __syncthreads();

  // layer 1: [128,64] @ Wx0[256,64]^T  (K = 64)
  v16bf a0 = load_afrag(xs, XD, m0, 0, lane);
  v16bf a1 = load_afrag(xs, XD, m0, 32, lane);
  for (int nt = 0; nt < 16; ++nt) {
    v8f acc = {};
    acc = wmma_bf(a0, load_bfrag(Wx0b, XD, nt * 16, 0, lane), acc);
    acc = wmma_bf(a1, load_bfrag(Wx0b, XD, nt * 16, 32, lane), acc);
    const int n = nt * 16 + n_l;
    const float b = bx0[n];
    for (int i = 0; i < 8; ++i)
      act1[(m0 + rb + i) * F_DIM + n] = f2bf(fast_tanh(acc[i] + b));
  }
  // rows of act1 are wave-private: in-wave LDS ordering suffices, no barrier

  // layer 2: [128,256] @ Wx1[256,256]^T  (K = 256)
  v16bf af[8];
  for (int kt = 0; kt < 8; ++kt) af[kt] = load_afrag(act1, F_DIM, m0, kt * 32, lane);
  for (int nt = 0; nt < 16; ++nt) {
    v8f acc = {};
    for (int kt = 0; kt < 8; ++kt)
      acc = wmma_bf(af[kt], load_bfrag(Wx1b, F_DIM, nt * 16, kt * 32, lane), acc);
    const int n = nt * 16 + n_l;
    const float b = bx1[n];
    for (int i = 0; i < 8; ++i)
      featb[(rowBase + m0 + rb + i) * F_DIM + n] = f2bf(fast_tanh(acc[i] + b));
  }
}

// --------------------------- stage 2: leaky-RNN scan ------------------------
// 16 blocks; block b owns batch rows [16b, 16b+16). h lives in LDS (fp32
// master double-buffer + bf16 double-buffer). feat tiles are double-buffered
// into LDS with async-to-LDS loads issued one step ahead. Writes bf16 h_t
// history (pre-update state) to global for stage 3.
__global__ __launch_bounds__(256) void scan_kernel(
    const unsigned short* __restrict__ featb,
    const unsigned short* __restrict__ Wihb,
    const unsigned short* __restrict__ Whhb,
    const float* __restrict__ bih, const float* __restrict__ bhh,
    unsigned short* __restrict__ hhist) {
  __shared__ __align__(16) unsigned short hb[2][16 * H_DIM];  // bf16 h
  __shared__ __align__(16) float hf[2][16 * H_DIM];           // fp32 master h
  __shared__ __align__(16) unsigned short fb[2][16 * F_DIM];  // feat tiles

  const int tid = threadIdx.x, lane = tid & 31, wave = tid >> 5;
  const int m0 = blockIdx.x * 16;  // global batch-row tile
  const int n_l = lane & 15, rb = (lane >> 4) << 3;
  const size_t tileBytes = (size_t)16 * F_DIM * 2;  // 8 KB, 512 x b128

  for (int i = tid; i < 16 * H_DIM; i += 256) {
    hb[0][i] = 0;
    hf[0][i] = 0.0f;
  }
  // prologue: async-stage feat tile for t = 0 into fb[0]
  {
    const char* g0 = (const char*)(featb + (size_t)m0 * F_DIM);
    const unsigned d0 = lds_off(&fb[0][0]);
    for (int c = tid; c < (int)(tileBytes / 16); c += 256)
      async_ld_b128(d0 + c * 16, g0 + (size_t)c * 16);
  }

  for (int t = 0; t < T_LEN; ++t) {
    const int cur = t & 1, nxt = cur ^ 1;
    unsigned short* hn = hhist + (size_t)(t + 1) * (B_DIM * H_DIM);

    wait_async0();    // feat tile t resident in fb[cur] (per-wave loads done)
    __syncthreads();  // ... across all waves; also protects hb/hf from t-1

    v16bf hA[8], fA[8];
    for (int kt = 0; kt < 8; ++kt) {
      hA[kt] = load_afrag(hb[cur], H_DIM, 0, kt * 32, lane);
      fA[kt] = load_afrag(fb[cur], F_DIM, 0, kt * 32, lane);
    }

    // overlap: async-stage feat tile t+1 into fb[nxt] behind this step's WMMAs
    if (t + 1 < T_LEN) {
      const char* g1 = (const char*)(featb + (size_t)(t + 1) * (B_DIM * F_DIM) +
                                     (size_t)m0 * F_DIM);
      const unsigned d1 = lds_off(&fb[nxt][0]);
      for (int c = tid; c < (int)(tileBytes / 16); c += 256)
        async_ld_b128(d1 + c * 16, g1 + (size_t)c * 16);
    }
    if (t + 2 < T_LEN)  // warm L2 two steps ahead
      __builtin_prefetch(featb + (size_t)(t + 2) * (B_DIM * F_DIM) +
                             (size_t)m0 * F_DIM + tid * 16,
                         0, 1);

    for (int j = 0; j < 2; ++j) {
      const int n0 = (wave * 2 + j) * 16;
      v8f acc = {};
      for (int kt = 0; kt < 8; ++kt)
        acc = wmma_bf(fA[kt], load_bfrag(Wihb, F_DIM, n0, kt * 32, lane), acc);
      for (int kt = 0; kt < 8; ++kt)
        acc = wmma_bf(hA[kt], load_bfrag(Whhb, H_DIM, n0, kt * 32, lane), acc);
      const int u = n_l + n0;  // hidden-unit index
      const float bb = bih[u] + bhh[u];
      // sorted alpha assignment: 86x0.001, 85x0.01, 85x0.1
      const float alpha = (u < 86) ? 0.001f : ((u < 171) ? 0.01f : 0.1f);
      for (int i = 0; i < 8; ++i) {
        const int r = rb + i;  // local batch row
        const float hv = hf[cur][r * H_DIM + u];
        const float hraw = fast_tanh(acc[i] + bb);
        const float hnew = (1.0f - alpha) * hv + alpha * hraw;
        hf[nxt][r * H_DIM + u] = hnew;
        const unsigned short h16 = f2bf(hnew);
        hb[nxt][r * H_DIM + u] = h16;
        hn[(size_t)(m0 + r) * H_DIM + u] = h16;  // history for stage 3
      }
    }
    __syncthreads();  // all waves done with fb[cur]/hb[cur] before next step
  }
}

// --------------------------- stage 3: output MLP ----------------------------
// y_t = (tanh(tanh(h_t@Wh0^T+bh0)@Wh1^T+bh1))@Wg^T + bg, parallel over (t,b).
// The block's 128x256 bf16 h rows are async-staged into LDS first.
__global__ __launch_bounds__(256) void mlp_kernel(
    const unsigned short* __restrict__ hhist,
    const unsigned short* __restrict__ Wh0b, const float* __restrict__ bh0,
    const unsigned short* __restrict__ Wh1b, const float* __restrict__ bh1,
    const unsigned short* __restrict__ Wgb, const float* __restrict__ bg,
    float* __restrict__ out) {
  extern __shared__ __align__(16) unsigned short smem[];
  unsigned short* a1 = smem;                  // [128][256] bf16
  unsigned short* a2 = smem + 128 * F_DIM;    // [128][256] bf16
  unsigned short* hs = smem + 2 * 128 * F_DIM;// [128][256] bf16 staged h

  const int tid = threadIdx.x, lane = tid & 31, wave = tid >> 5;
  const size_t rowBase = (size_t)blockIdx.x * 128;  // rows of [T*B]
  const int m0 = wave * 16;
  const int n_l = lane & 15, rb = (lane >> 4) << 3;

  // async-stage 64 KB of h rows (bf16, contiguous) into LDS
  {
    const char* g = (const char*)(hhist + rowBase * (size_t)H_DIM);
    const unsigned d = lds_off(hs);
    for (int c = tid; c < (128 * H_DIM * 2) / 16; c += 256)  // 4096 x b128
      async_ld_b128(d + c * 16, g + (size_t)c * 16);
  }
  wait_async0();
  __syncthreads();

  // layer 1: h @ Wh0^T
  v16bf af[8];
  for (int kt = 0; kt < 8; ++kt) af[kt] = load_afrag(hs, H_DIM, m0, kt * 32, lane);
  for (int nt = 0; nt < 16; ++nt) {
    v8f acc = {};
    for (int kt = 0; kt < 8; ++kt)
      acc = wmma_bf(af[kt], load_bfrag(Wh0b, H_DIM, nt * 16, kt * 32, lane), acc);
    const int n = nt * 16 + n_l;
    const float b = bh0[n];
    for (int i = 0; i < 8; ++i)
      a1[(m0 + rb + i) * F_DIM + n] = f2bf(fast_tanh(acc[i] + b));
  }

  // layer 2: d1 @ Wh1^T   (rows are wave-private; in-wave LDS ordering OK)
  v16bf bf[8];
  for (int kt = 0; kt < 8; ++kt) bf[kt] = load_afrag(a1, F_DIM, m0, kt * 32, lane);
  for (int nt = 0; nt < 16; ++nt) {
    v8f acc = {};
    for (int kt = 0; kt < 8; ++kt)
      acc = wmma_bf(bf[kt], load_bfrag(Wh1b, F_DIM, nt * 16, kt * 32, lane), acc);
    const int n = nt * 16 + n_l;
    const float b = bh1[n];
    for (int i = 0; i < 8; ++i)
      a2[(m0 + rb + i) * F_DIM + n] = f2bf(fast_tanh(acc[i] + b));
  }

  // layer 3: d2 @ Wg^T + bg  -> y (fp32)
  v16bf cf[8];
  for (int kt = 0; kt < 8; ++kt) cf[kt] = load_afrag(a2, F_DIM, m0, kt * 32, lane);
  for (int nt = 0; nt < 4; ++nt) {
    v8f acc = {};
    for (int kt = 0; kt < 8; ++kt)
      acc = wmma_bf(cf[kt], load_bfrag(Wgb, F_DIM, nt * 16, kt * 32, lane), acc);
    const int n = nt * 16 + n_l;
    const float b = bg[n];
    for (int i = 0; i < 8; ++i)
      out[(rowBase + m0 + rb + i) * XD + n] = acc[i] + b;
  }
}

// ---------------------------------------------------------------------------
extern "C" void kernel_launch(void* const* d_in, const int* in_sizes, int n_in,
                              void* d_out, int out_size, void* d_ws,
                              size_t ws_size, hipStream_t stream) {
  (void)in_sizes; (void)n_in; (void)out_size; (void)ws_size;

  const float* x   = (const float*)d_in[0];
  const float* Wx0 = (const float*)d_in[1];
  const float* bx0 = (const float*)d_in[2];
  const float* Wx1 = (const float*)d_in[3];
  const float* bx1 = (const float*)d_in[4];
  const float* Wih = (const float*)d_in[5];
  const float* Whh = (const float*)d_in[6];
  const float* bih = (const float*)d_in[7];
  const float* bhh = (const float*)d_in[8];
  const float* Wh0 = (const float*)d_in[9];
  const float* bh0 = (const float*)d_in[10];
  const float* Wh1 = (const float*)d_in[11];
  const float* bh1 = (const float*)d_in[12];
  const float* Wg  = (const float*)d_in[13];
  const float* bg  = (const float*)d_in[14];
  float* out = (float*)d_out;

  char* ws = (char*)d_ws;
  size_t o = 0;
  auto alloc = [&](size_t bytes) {
    void* p = ws + o;
    o = (o + bytes + 255) & ~(size_t)255;
    return p;
  };
  unsigned short* Wx0b = (unsigned short*)alloc((size_t)F_DIM * XD * 2);
  unsigned short* Wx1b = (unsigned short*)alloc((size_t)F_DIM * F_DIM * 2);
  unsigned short* Wihb = (unsigned short*)alloc((size_t)H_DIM * F_DIM * 2);
  unsigned short* Whhb = (unsigned short*)alloc((size_t)H_DIM * H_DIM * 2);
  unsigned short* Wh0b = (unsigned short*)alloc((size_t)F_DIM * H_DIM * 2);
  unsigned short* Wh1b = (unsigned short*)alloc((size_t)F_DIM * F_DIM * 2);
  unsigned short* Wgb  = (unsigned short*)alloc((size_t)XD * F_DIM * 2);
  unsigned short* featb =
      (unsigned short*)alloc((size_t)T_LEN * B_DIM * F_DIM * 2);
  unsigned short* hhist =
      (unsigned short*)alloc((size_t)(T_LEN + 1) * B_DIM * H_DIM * 2);

  // fp32 -> bf16 weight conversion (deterministic, every launch)
  cvt_kernel<<<(F_DIM * XD + 255) / 256, 256, 0, stream>>>(Wx0, Wx0b, F_DIM * XD);
  cvt_kernel<<<(F_DIM * F_DIM + 255) / 256, 256, 0, stream>>>(Wx1, Wx1b, F_DIM * F_DIM);
  cvt_kernel<<<(H_DIM * F_DIM + 255) / 256, 256, 0, stream>>>(Wih, Wihb, H_DIM * F_DIM);
  cvt_kernel<<<(H_DIM * H_DIM + 255) / 256, 256, 0, stream>>>(Whh, Whhb, H_DIM * H_DIM);
  cvt_kernel<<<(F_DIM * H_DIM + 255) / 256, 256, 0, stream>>>(Wh0, Wh0b, F_DIM * H_DIM);
  cvt_kernel<<<(F_DIM * F_DIM + 255) / 256, 256, 0, stream>>>(Wh1, Wh1b, F_DIM * F_DIM);
  cvt_kernel<<<(XD * F_DIM + 255) / 256, 256, 0, stream>>>(Wg, Wgb, XD * F_DIM);
  // h_0 = 0
  zero16_kernel<<<(B_DIM * H_DIM + 255) / 256, 256, 0, stream>>>(hhist,
                                                                 B_DIM * H_DIM);

  const size_t featSmem = (size_t)(128 * XD + 128 * F_DIM) * 2;   //  80 KB
  const size_t mlpSmem  = (size_t)(3 * 128 * F_DIM) * 2;          // 192 KB

  feat_kernel<<<(T_LEN * B_DIM) / 128, 256, featSmem, stream>>>(
      x, Wx0b, bx0, Wx1b, bx1, featb);
  scan_kernel<<<B_DIM / 16, 256, 0, stream>>>(featb, Wihb, Whhb, bih, bhh,
                                              hhist);
  mlp_kernel<<<(T_LEN * B_DIM) / 128, 256, mlpSmem, stream>>>(
      hhist, Wh0b, bh0, Wh1b, bh1, Wgb, bg, out);
}
